// RNNLayer_40879498728894
// MI455X (gfx1250) — matrix operand order repaired and checked by reading, not compiled, not run on hardware
//
#include <hip/hip_runtime.h>
#include <hip/hip_bf16.h>
#include <stdint.h>

// ---------------------------------------------------------------------------
// LSTM layer, B=64, T=128, F=H=1024, TF gate order i,j,f,o, forget bias 1.0
//   Phase 1: X_proj[b*T+t, 4H] = x[b,t,:] @ W[0:F, :] + bias      (parallel)
//   Phase 2: 128 sequential steps:
//            z = X_proj[t] + h_{t-1} @ W[F:F+H, :]
//            c = c*sig(zf+1) + sig(zi)*tanh(zj);  h = tanh(c)*sig(zo)
// All GEMMs: v_wmma_f32_16x16x32_bf16 (bf16 in, f32 accumulate).
// Cache policy: Wh (8 MB bf16) kept L2-resident via WGP-scope prefetch;
// single-touch streams (x, X_proj, out) use non-temporal hints.
// ---------------------------------------------------------------------------

#define Bsz   64
#define Tsz   128
#define Fsz   1024
#define Hsz   1024
#define N4H   4096          // 4*H
#define KTOT  1024          // K of both GEMMs
#define NTN   (N4H / 16)    // 256 column tiles
#define NTK   (KTOT / 32)   // 32 K-chunks

typedef __attribute__((ext_vector_type(16))) __bf16 v16bf;
typedef __attribute__((ext_vector_type(8)))  __bf16 v8bf;
typedef __attribute__((ext_vector_type(8)))  float  v8f;

static __device__ __forceinline__ unsigned short f32_to_bf16(float f) {
    unsigned int u = __float_as_uint(f);
    unsigned int r = 0x7FFFu + ((u >> 16) & 1u);   // round-to-nearest-even
    return (unsigned short)((u + r) >> 16);
}

static __device__ __forceinline__ float sigf(float x) {
    return 1.0f / (1.0f + __expf(-x));
}

// tanh(x) = 2*sigmoid(2x) - 1, built on the __expf transcendental
static __device__ __forceinline__ float tanh_fast(float x) {
    return __builtin_fmaf(2.0f, sigf(2.0f * x), -1.0f);
}

// A fragment (16x32 bf16) from a row-major bf16 matrix with leading dim ldk.
// ISA layout: lanes 0-15 hold K {0..7,16..23}, lanes 16-31 hold K {8..15,24..31}.
static __device__ __forceinline__ v16bf load_a_frag(const unsigned short* A,
                                                    int ldk, int row0, int kbase,
                                                    int lane) {
    int r  = row0 + (lane & 15);
    int hl = lane >> 4;
    const __bf16* p = reinterpret_cast<const __bf16*>(A) +
                      (size_t)r * ldk + kbase + hl * 8;
    v8bf lo = *reinterpret_cast<const v8bf*>(p);        // K base+hl*8 .. +7
    v8bf hi = *reinterpret_cast<const v8bf*>(p + 16);   // K base+16+hl*8 .. +7
    return __builtin_shufflevector(lo, hi, 0,1,2,3,4,5,6,7,
                                           8,9,10,11,12,13,14,15);
}

// B fragment (32x16 bf16) from pre-swizzled weights: tile (kt,nt) stored as
// 512 contiguous bf16 in exact per-lane fragment order (lane*16 + e).
static __device__ __forceinline__ v16bf load_b_frag(const unsigned short* Wp,
                                                    int kt, int nt, int lane) {
    const __bf16* p = reinterpret_cast<const __bf16*>(Wp) +
                      ((size_t)(kt * NTN + nt) * 512) + lane * 16;
    return *reinterpret_cast<const v16bf*>(p);          // one 32-byte load
}

static __device__ __forceinline__ v8f wmma_bf16(v16bf a, v16bf b, v8f c) {
    return __builtin_amdgcn_wmma_f32_16x16x32_bf16(
        /*neg_a=*/false, a, /*neg_b=*/false, b,
        /*c_mod=*/(short)0, c, /*reuse_a=*/false, /*reuse_b=*/false);
}

// ---------------------------------------------------------------- pack / cvt
__global__ void k_cvt_x(const float* __restrict__ x,
                        unsigned short* __restrict__ xb, int n) {
    int i = blockIdx.x * blockDim.x + threadIdx.x;
    if (i < n) xb[i] = f32_to_bf16(__builtin_nontemporal_load(x + i));
}

// Swizzle W [(F+H) x 4H] fp32 row-major -> bf16 B-fragment tile order.
// Element (mat, kt, nt, lane, e):  srcK = kt*32 + (lane/16)*8 + (e%8) + (e/8)*16,
//                                  srcN = nt*16 + (lane%16)
__global__ void k_pack_w(const float* __restrict__ W,
                         unsigned short* __restrict__ Wxp,
                         unsigned short* __restrict__ Whp) {
    int idx = blockIdx.x * blockDim.x + threadIdx.x;          // < 2*4M
    if (idx >= 2 * NTK * NTN * 512) return;
    int e    =  idx        & 15;
    int lane = (idx >> 4)  & 31;
    int nt   = (idx >> 9)  & (NTN - 1);
    int kt   = (idx >> 17) & (NTK - 1);
    int mat  =  idx >> 22;                                    // 0=Wx, 1=Wh
    int srcK = kt * 32 + (lane >> 4) * 8 + (e & 7) + (e >> 3) * 16;
    int srcN = nt * 16 + (lane & 15);
    float v  = __builtin_nontemporal_load(
                   W + (size_t)(mat * Fsz + srcK) * N4H + srcN);
    unsigned short* dst = mat ? Whp : Wxp;
    dst[idx & (NTK * NTN * 512 - 1)] = f32_to_bf16(v);
}

__global__ void k_init_state(float* __restrict__ c,
                             unsigned short* __restrict__ h0, int n) {
    int i = blockIdx.x * blockDim.x + threadIdx.x;
    if (i < n) { c[i] = 0.0f; h0[i] = 0; }
}

// ------------------------------------------------------------ X projection
// X_proj = x_bf16 @ Wx + bias.  grid(64, 128): x = N/64 chunk, y = M/64 chunk.
__global__ void __launch_bounds__(256)
k_xproj(const unsigned short* __restrict__ xb,
        const unsigned short* __restrict__ Wxp,
        const float* __restrict__ bias,
        float* __restrict__ xproj) {
    const int lane   = threadIdx.x & 31;
    const int wave   = threadIdx.x >> 5;
    const int m_base = blockIdx.y * 64 + (wave & 3) * 16;
    const int n_base = blockIdx.x * 64 + (wave >> 2) * 32;
    const int nt0    = n_base >> 4;

    v8f acc0 = {}; v8f acc1 = {};
    #pragma unroll 8
    for (int kc = 0; kc < NTK; ++kc) {
        if (kc + 1 < NTK)   // WGP-scope prefetch of next weight chunk
            __builtin_prefetch(reinterpret_cast<const __bf16*>(Wxp) +
                               ((size_t)((kc + 1) * NTN + nt0) * 512), 0, 3);
        v16bf a  = load_a_frag(xb, KTOT, m_base, kc * 32, lane);
        v16bf b0 = load_b_frag(Wxp, kc, nt0,     lane);
        v16bf b1 = load_b_frag(Wxp, kc, nt0 + 1, lane);
        acc0 = wmma_bf16(a, b0, acc0);
        acc1 = wmma_bf16(a, b1, acc1);
    }
    const int   colL = lane & 15;
    const float bv0  = bias[n_base + colL];
    const float bv1  = bias[n_base + 16 + colL];
    #pragma unroll
    for (int r = 0; r < 8; ++r) {
        int row = m_base + r + (lane >> 4) * 8;               // C/D layout
        size_t o = (size_t)row * N4H + n_base + colL;
        __builtin_nontemporal_store(acc0[r] + bv0, xproj + o);       // stream
        __builtin_nontemporal_store(acc1[r] + bv1, xproj + o + 16);  // stream
    }
}

// ---------------------------------------------------------------- LSTM step
// Block wg owns hidden cols [16*wg, 16*wg+16) in ALL four gates.
// Waves 0-3: gates {i,j} for m-tiles 0-3; waves 4-7: gates {f,o}.
__global__ void __launch_bounds__(256)
k_step(const unsigned short* __restrict__ hprev,
       const unsigned short* __restrict__ Whp,
       const float* __restrict__ xproj,
       float* __restrict__ c_state,
       unsigned short* __restrict__ hnext,
       float* __restrict__ out, int t) {
    __shared__ float zt[4][64][16];                           // 16 KB of 320 KB

    const int lane   = threadIdx.x & 31;
    const int wave   = threadIdx.x >> 5;
    const int wg     = blockIdx.x;                            // 0..63
    const int n0     = wg * 16;
    const int m_base = (wave & 3) * 16;
    const int g0     = (wave >> 2) * 2;                       // gate pair
    const int nt0    = g0 * 64 + wg;                          // gate g0 tile
    const int nt1    = (g0 + 1) * 64 + wg;                    // gate g0+1 tile

    v8f acc0 = {}; v8f acc1 = {};
    #pragma unroll 8
    for (int kc = 0; kc < NTK; ++kc) {
        if (kc + 1 < NTK)   // keep Wh hot at WGP scope
            __builtin_prefetch(reinterpret_cast<const __bf16*>(Whp) +
                               ((size_t)((kc + 1) * NTN + nt0) * 512), 0, 3);
        v16bf a  = load_a_frag(hprev, Hsz, m_base, kc * 32, lane);
        v16bf b0 = load_b_frag(Whp, kc, nt0, lane);
        v16bf b1 = load_b_frag(Whp, kc, nt1, lane);
        acc0 = wmma_bf16(a, b0, acc0);
        acc1 = wmma_bf16(a, b1, acc1);
    }

    // stage z = h@Wh + X_proj into LDS, grouped by gate (X_proj: single read)
    const int colL = lane & 15;
    #pragma unroll
    for (int r = 0; r < 8; ++r) {
        int rowt = r + (lane >> 4) * 8;
        int b    = m_base + rowt;                             // batch index
        size_t xrow = ((size_t)b * Tsz + t) * N4H;
        float xp0 = __builtin_nontemporal_load(
                        xproj + xrow + (size_t)g0 * Hsz + n0 + colL);
        float xp1 = __builtin_nontemporal_load(
                        xproj + xrow + (size_t)(g0 + 1) * Hsz + n0 + colL);
        zt[g0    ][b][colL] = acc0[r] + xp0;
        zt[g0 + 1][b][colL] = acc1[r] + xp1;
    }
    __syncthreads();

    // fused gate nonlinearity + state update (c slice exclusive to this block)
    for (int it = threadIdx.x; it < Bsz * 16; it += 256) {
        int b   = it >> 4;
        int col = it & 15;
        float zi = zt[0][b][col];
        float zj = zt[1][b][col];
        float zf = zt[2][b][col];
        float zo = zt[3][b][col];
        int   ci = b * Hsz + n0 + col;
        float c_old = c_state[ci];
        float c_new = c_old * sigf(zf + 1.0f) + sigf(zi) * tanh_fast(zj);
        float h_new = tanh_fast(c_new) * sigf(zo);
        c_state[ci] = c_new;
        hnext[ci]   = f32_to_bf16(h_new);
        __builtin_nontemporal_store(
            h_new, out + ((size_t)b * Tsz + t) * Hsz + n0 + col);
    }
}

// --------------------------------------------------------------------- host
extern "C" void kernel_launch(void* const* d_in, const int* in_sizes, int n_in,
                              void* d_out, int out_size, void* d_ws, size_t ws_size,
                              hipStream_t stream) {
    const float* x    = (const float*)d_in[0];   // [B,T,F]
    const float* W    = (const float*)d_in[1];   // [F+H, 4H]
    const float* bias = (const float*)d_in[2];   // [4H]
    float*       out  = (float*)d_out;           // [B,T,H]

    // workspace layout (256B aligned chunks)
    uint8_t* ws = (uint8_t*)d_ws;
    size_t off = 0;
    auto take = [&](size_t bytes) {
        uint8_t* p = ws + off;
        off += (bytes + 255) & ~(size_t)255;
        return p;
    };
    unsigned short* xb   = (unsigned short*)take((size_t)Bsz*Tsz*Fsz*2);   // 16 MB
    unsigned short* Wxp  = (unsigned short*)take((size_t)KTOT*N4H*2);      //  8 MB
    unsigned short* Whp  = (unsigned short*)take((size_t)KTOT*N4H*2);      //  8 MB
    float*          xprj = (float*)take((size_t)Bsz*Tsz*N4H*4);            // 128 MB
    float*          cst  = (float*)take((size_t)Bsz*Hsz*4);
    unsigned short* h0   = (unsigned short*)take((size_t)Bsz*Hsz*2);
    unsigned short* h1   = (unsigned short*)take((size_t)Bsz*Hsz*2);
    (void)ws_size; (void)in_sizes; (void)n_in; (void)out_size;

    const int nx = Bsz * Tsz * Fsz;
    k_cvt_x<<<(nx + 255) / 256, 256, 0, stream>>>(x, xb, nx);

    const int nw = 2 * NTK * NTN * 512;
    k_pack_w<<<(nw + 255) / 256, 256, 0, stream>>>(W, Wxp, Whp);

    const int ns = Bsz * Hsz;
    k_init_state<<<(ns + 255) / 256, 256, 0, stream>>>(cst, h0, ns);

    // Phase 1: bulk input projection (half of all FLOPs, fully parallel)
    k_xproj<<<dim3(N4H / 64, (Bsz * Tsz) / 64), 256, 0, stream>>>(xb, Wxp, bias, xprj);

    // Phase 2: 128 sequential recurrent steps (Wh stays L2-resident)
    for (int t = 0; t < Tsz; ++t) {
        const unsigned short* hp = (t & 1) ? h1 : h0;
        unsigned short*       hn = (t & 1) ? h0 : h1;
        k_step<<<Hsz / 16, 256, 0, stream>>>(hp, Whp, xprj, cst, hn, out, t);
    }
}
